// Block_Group_27685359190798
// MI455X (gfx1250) — matrix operand order repaired and checked by reading, not compiled, not run on hardware
//
#include <hip/hip_runtime.h>
#include <math.h>

// Problem dims
#define BATCH   8
#define LL      4096            // 64*64
#define DIMM    256
#define MTOT    (BATCH*LL)      // 32768 rows
#define KDIR    4
#define DGRP    64
#define NST     16
#define RRANK   4
#define CDBL    36              // R + 2N

typedef __attribute__((ext_vector_type(16))) _Float16 v16h;
typedef __attribute__((ext_vector_type(8)))  float    v8f;

// ---------------- WMMA fragment loaders (wave32; k = 16*(lane/16) + s in both A and B) ----

// A 16x32 f16 from row-major _Float16 [m, k]
__device__ __forceinline__ v16h load_fragA_rm_f16(const _Float16* base, int ld, int lane) {
  int r  = lane & 15;
  int kb = (lane >> 4) * 16;            // lanes<16: K 0..15 ; lanes>=16: 16..31
  const _Float16* p = base + (size_t)r * ld;
  v16h f;
#pragma unroll
  for (int i = 0; i < 8; ++i) {
    f[i]     = p[kb + i];
    f[i + 8] = p[kb + 8 + i];
  }
  return f;
}

// A 16x32 f16 from k-major f32: element (m,k) = base[k*ldk + m]
__device__ __forceinline__ v16h load_fragA_cm_f32(const float* base, int ldk, int lane) {
  int r  = lane & 15;
  int kb = (lane >> 4) * 16;
  v16h f;
#pragma unroll
  for (int i = 0; i < 8; ++i) {
    f[i]     = (_Float16)base[(size_t)(kb + i) * ldk + r];
    f[i + 8] = (_Float16)base[(size_t)(kb + 8 + i) * ldk + r];
  }
  return f;
}

// B 32x16 f16 from [N,K] row-major f16: lanes<16 N=lane K0..15, lanes>=16 K16..31
__device__ __forceinline__ v16h load_fragB_nk_f16(const _Float16* base, int ld, int lane) {
  int nn = lane & 15;
  int kb = (lane >> 4) * 16;
  const _Float16* p = base + (size_t)nn * ld + kb;
  v16h f;
#pragma unroll
  for (int i = 0; i < 16; ++i) f[i] = p[i];
  return f;
}

// Same but zero-fills rows n >= nmax (for N=36 padded to 48)
__device__ __forceinline__ v16h load_fragB_nk_f16_guard(const _Float16* base, int ld,
                                                        int n0, int nmax, int lane) {
  int nn = n0 + (lane & 15);
  int kb = (lane >> 4) * 16;
  v16h f;
  if (nn < nmax) {
    const _Float16* p = base + (size_t)nn * ld + kb;
#pragma unroll
    for (int i = 0; i < 16; ++i) f[i] = p[i];
  } else {
#pragma unroll
    for (int i = 0; i < 16; ++i) f[i] = (_Float16)0.f;
  }
  return f;
}

// ---------------- Utility kernels ---------------------------------------------

__global__ void k_zero(float* p) { p[blockIdx.x * 256 + threadIdx.x] = 0.f; }

__global__ void k_cvt(const float* __restrict__ src, _Float16* __restrict__ dst, int n) {
  int i = blockIdx.x * blockDim.x + threadIdx.x;
  if (i < n) dst[i] = (_Float16)src[i];
}

// ---------------- GEMM kernels ------------------------------------------------

// GEMM1: xz[m, n] = sum_k x[m,k] * Win[n,k]   (M=32768, N=512, K=256)
// one wave computes a 16x128 strip; all 8 B fragments are materialized per K-step
// so their global_load_b128s issue as one clause and overlap the WMMA chain.
__global__ void k_gemm_inproj(const _Float16* __restrict__ x16, const _Float16* __restrict__ w16,
                              float* __restrict__ xp, float* __restrict__ z) {
  int wave = blockIdx.x * (blockDim.x >> 5) + (threadIdx.x >> 5);   // 0..8191
  int mt  = wave >> 2;
  int ntg = wave & 3;                 // 4 strips of 128 columns
  int m0 = mt * 16;
  int lane = threadIdx.x & 31;
  v8f acc[8] = {};
#pragma unroll
  for (int k0 = 0; k0 < 256; k0 += 32) {
    v16h a = load_fragA_rm_f16(x16 + (size_t)m0 * 256 + k0, 256, lane);
    v16h bfr[8];
#pragma unroll
    for (int nt = 0; nt < 8; ++nt)
      bfr[nt] = load_fragB_nk_f16(w16 + (size_t)(ntg * 128 + nt * 16) * 256 + k0, 256, lane);
#pragma unroll
    for (int nt = 0; nt < 8; ++nt)
      acc[nt] = __builtin_amdgcn_wmma_f32_16x16x32_f16(false, a, false, bfr[nt], (short)0,
                                                       acc[nt], false, false);
  }
  int n = lane & 15, sub = lane >> 4;
  float* dst  = (ntg < 2) ? xp : z;
  int   nbase = (ntg < 2) ? ntg * 128 : (ntg - 2) * 128;
#pragma unroll
  for (int nt = 0; nt < 8; ++nt)
#pragma unroll
    for (int v = 0; v < 8; ++v)
      dst[(size_t)(m0 + v + sub * 8) * 256 + nbase + nt * 16 + n] = acc[nt][v];
}

// GEMM2: x_dbl[b,k,c,l] = sum_dg xs[b,k,dg,l] * xpw[k,c,dg]  (per k: M=32768, N=36->48, K=64)
// one wave computes all 3 N-tiles -> strided A read exactly once.
__global__ void k_gemm_xdbl(const float* __restrict__ xs, const _Float16* __restrict__ xpw16,
                            float* __restrict__ xdbl) {
  int wave = blockIdx.x * (blockDim.x >> 5) + (threadIdx.x >> 5);   // 0..8191
  int mt = wave & 2047;
  int k  = wave >> 11;
  int m0 = mt * 16;
  int b  = m0 >> 12;                  // tiles never straddle batches (4096 % 16 == 0)
  int l0 = m0 & 4095;
  int lane = threadIdx.x & 31;
  const float*    abase = xs + (((size_t)b * 4 + k) * DGRP) * LL + l0; // A[m,dg]=abase[dg*LL+m]
  const _Float16* bbase = xpw16 + (size_t)k * CDBL * DGRP;             // B[n,dg]=bbase[n*64+dg]
  v8f acc[3] = {};
#pragma unroll
  for (int k0 = 0; k0 < 64; k0 += 32) {
    v16h a = load_fragA_cm_f32(abase + (size_t)k0 * LL, LL, lane);
    v16h bfr[3];
#pragma unroll
    for (int nt = 0; nt < 3; ++nt)
      bfr[nt] = load_fragB_nk_f16_guard(bbase + k0, DGRP, nt * 16, CDBL, lane);
#pragma unroll
    for (int nt = 0; nt < 3; ++nt)
      acc[nt] = __builtin_amdgcn_wmma_f32_16x16x32_f16(false, a, false, bfr[nt], (short)0,
                                                       acc[nt], false, false);
  }
  int sub = lane >> 4;
#pragma unroll
  for (int nt = 0; nt < 3; ++nt) {
    int n = nt * 16 + (lane & 15);
    if (n < CDBL) {
#pragma unroll
      for (int v = 0; v < 8; ++v)
        xdbl[(((size_t)b * 4 + k) * CDBL + n) * LL + l0 + v + sub * 8] = acc[nt][v];
    }
  }
}

// GEMM3: out[m, o] = sum_c yln[m,c] * Wout[o,c]  (M=32768, N=256, K=256), 16x128 per wave
__global__ void k_gemm_out(const _Float16* __restrict__ yln, const _Float16* __restrict__ w16,
                           float* __restrict__ out) {
  int wave = blockIdx.x * (blockDim.x >> 5) + (threadIdx.x >> 5);   // 0..4095
  int mt  = wave >> 1;
  int ntg = wave & 1;
  int m0 = mt * 16;
  int lane = threadIdx.x & 31;
  v8f acc[8] = {};
#pragma unroll
  for (int k0 = 0; k0 < 256; k0 += 32) {
    v16h a = load_fragA_rm_f16(yln + (size_t)m0 * 256 + k0, 256, lane);
    v16h bfr[8];
#pragma unroll
    for (int nt = 0; nt < 8; ++nt)
      bfr[nt] = load_fragB_nk_f16(w16 + (size_t)(ntg * 128 + nt * 16) * 256 + k0, 256, lane);
#pragma unroll
    for (int nt = 0; nt < 8; ++nt)
      acc[nt] = __builtin_amdgcn_wmma_f32_16x16x32_f16(false, a, false, bfr[nt], (short)0,
                                                       acc[nt], false, false);
  }
  int n = lane & 15, sub = lane >> 4;
#pragma unroll
  for (int nt = 0; nt < 8; ++nt)
#pragma unroll
    for (int v = 0; v < 8; ++v)
      out[(size_t)(m0 + v + sub * 8) * 256 + ntg * 128 + nt * 16 + n] = acc[nt][v];
}

// ---------------- Elementwise / scan kernels ----------------------------------

// silu(z) in place; xc = silu(conv_w*xp+conv_b); scatter 4-direction xs; SE partial sums
__global__ void k_elem(const float* __restrict__ xp, float* __restrict__ z,
                       const float* __restrict__ conv_w, const float* __restrict__ conv_b,
                       float* __restrict__ xs, float* __restrict__ zz) {
  int c = threadIdx.x;
  int b = blockIdx.x >> 8;            // 256 chunks of 16 pixels per batch
  int chunk = blockIdx.x & 255;
  float cw = conv_w[c], cb = conv_b[c];
  int dir = c & 3, d = c >> 2;
  float acc = 0.f;
#pragma unroll 4
  for (int i = 0; i < 16; ++i) {
    int p = chunk * 16 + i;
    int hh = p >> 6, ww = p & 63;
    size_t m = (size_t)b * LL + p;
    float zv = z[m * 256 + c];
    z[m * 256 + c] = zv / (1.f + __expf(-zv));              // silu
    float xv = xp[m * 256 + c];
    float xc = cw * xv + cb;
    xc = xc / (1.f + __expf(-xc));                          // silu
    acc += xc;
    int l;
    if (dir == 0)      l = p;
    else if (dir == 1) l = ww * 64 + hh;
    else if (dir == 2) l = 4095 - p;
    else               l = 4095 - (ww * 64 + hh);
    xs[(((size_t)b * 4 + dir) * DGRP + d) * LL + l] = xc;
  }
  atomicAdd(&zz[b * 256 + c], acc);
}

// SE gate: fc2_out = sigmoid(relu(mean @ fc1^T) @ fc2^T)
__global__ void k_se(const float* __restrict__ zz,
                     const float* __restrict__ fc1_w, const float* __restrict__ fc1_b,
                     const float* __restrict__ fc2_w, const float* __restrict__ fc2_b,
                     float* __restrict__ fc2_out) {
  __shared__ float zm[256];
  __shared__ float hid[4];
  int b = blockIdx.x, c = threadIdx.x;
  zm[c] = zz[b * 256 + c] * (1.f / 4096.f);
  __syncthreads();
  if (c < 4) {
    float s = fc1_b[c];
    for (int i = 0; i < 256; ++i) s += zm[i] * fc1_w[c * 256 + i];
    hid[c] = s > 0.f ? s : 0.f;
  }
  __syncthreads();
  float s = fc2_b[c];
#pragma unroll
  for (int j = 0; j < 4; ++j) s += hid[j] * fc2_w[c * 4 + j];
  fc2_out[b * 256 + c] = 1.f / (1.f + __expf(-s));
}

// Selective scan: one lane per (dg,n) state; wave handles a dg pair of one (b,k).
// 1024 waves of independent exp/fma recurrences; 16-lane shuffle reduction for y.
// global_prefetch keeps the 7 input streams ahead of the serial dependence chain.
__global__ void k_scan(const float* __restrict__ xs, const float* __restrict__ xdbl,
                       const float* __restrict__ dtw, const float* __restrict__ dtb,
                       const float* __restrict__ alogs, const float* __restrict__ Dsp,
                       float* __restrict__ out_y) {
  int wave = blockIdx.x * (blockDim.x >> 5) + (threadIdx.x >> 5);   // 0..1023
  int lane = threadIdx.x & 31;
  int b   = wave >> 7;
  int rem = wave & 127;
  int k   = rem >> 5;
  int dg0 = (rem & 31) * 2;
  int sub = lane >> 4;
  int n   = lane & 15;
  int dg  = dg0 + sub;
  int kd  = k * DGRP + dg;
  float a    = -__expf(alogs[kd * NST + n]);
  float bias = dtb[kd];
  float Dc   = Dsp[kd];
  float w0 = dtw[kd * 4 + 0], w1 = dtw[kd * 4 + 1];
  float w2 = dtw[kd * 4 + 2], w3 = dtw[kd * 4 + 3];
  const float* xd = xdbl + ((size_t)(b * 4 + k) * CDBL) * LL;
  const float* Bp = xd + (size_t)(RRANK + n) * LL;
  const float* Cp = xd + (size_t)(RRANK + NST + n) * LL;
  const float* up = xs + ((size_t)(b * 4 + k) * DGRP + dg) * LL;
  float* yp = out_y + ((size_t)(b * 4 + k) * DGRP + dg) * LL;
  const int PF = 256;
  float h = 0.f;
  for (int t = 0; t < LL; ++t) {
    if ((t & 31) == 0 && t + PF < LL) {
      __builtin_prefetch(xd + t + PF, 0, 3);
      __builtin_prefetch(xd + LL + t + PF, 0, 3);
      __builtin_prefetch(xd + 2 * LL + t + PF, 0, 3);
      __builtin_prefetch(xd + 3 * LL + t + PF, 0, 3);
      __builtin_prefetch(Bp + t + PF, 0, 3);
      __builtin_prefetch(Cp + t + PF, 0, 3);
      __builtin_prefetch(up + t + PF, 0, 3);
    }
    float dtraw = xd[t] * w0 + xd[LL + t] * w1 + xd[2 * LL + t] * w2 + xd[3 * LL + t] * w3 + bias;
    float delta = dtraw > 20.f ? dtraw : log1pf(__expf(dtraw));     // softplus
    float u  = up[t];
    float Bv = Bp[t];
    float Cv = Cp[t];
    h = __expf(delta * a) * h + (delta * u) * Bv;
    float s = h * Cv;
    s += __shfl_xor(s, 1, 16);
    s += __shfl_xor(s, 2, 16);
    s += __shfl_xor(s, 4, 16);
    s += __shfl_xor(s, 8, 16);
    if (n == 0) yp[t] = s + u * Dc;
  }
}

// Recombine 4 directions, SE gate, LayerNorm(256), multiply silu(z), emit f16 rows
__global__ void k_ln(const float* __restrict__ out_y, const float* __restrict__ fc2_out,
                     const float* __restrict__ z, const float* __restrict__ ln_g,
                     const float* __restrict__ ln_b, _Float16* __restrict__ yln) {
  __shared__ float red[256];
  int m = blockIdx.x;                  // row 0..32767
  int c = threadIdx.x;
  int b = m >> 12;
  int p = m & 4095;
  int hh = p >> 6, ww = p & 63;
  int dir = c & 3, d = c >> 2;
  int l;
  if (dir == 0)      l = p;
  else if (dir == 1) l = ww * 64 + hh;
  else if (dir == 2) l = 4095 - p;
  else               l = 4095 - (ww * 64 + hh);
  float v = out_y[(((size_t)b * 4 + dir) * DGRP + d) * LL + l];
  v *= fc2_out[b * 256 + c];
  red[c] = v; __syncthreads();
  for (int s = 128; s > 0; s >>= 1) { if (c < s) red[c] += red[c + s]; __syncthreads(); }
  float mu = red[0] * (1.f / 256.f);
  __syncthreads();
  float dv = v - mu;
  red[c] = dv * dv; __syncthreads();
  for (int s = 128; s > 0; s >>= 1) { if (c < s) red[c] += red[c + s]; __syncthreads(); }
  float var = red[0] * (1.f / 256.f);
  float yv = dv * rsqrtf(var + 1e-5f) * ln_g[c] + ln_b[c];
  yv *= z[(size_t)m * 256 + c];
  yln[(size_t)m * 256 + c] = (_Float16)yv;
}

// ---------------- Launch ------------------------------------------------------

extern "C" void kernel_launch(void* const* d_in, const int* in_sizes, int n_in,
                              void* d_out, int out_size, void* d_ws, size_t ws_size,
                              hipStream_t stream) {
  const float* x      = (const float*)d_in[0];
  const float* w_in   = (const float*)d_in[1];
  const float* conv_w = (const float*)d_in[2];
  const float* conv_b = (const float*)d_in[3];
  const float* fc1_w  = (const float*)d_in[4];
  const float* fc1_b  = (const float*)d_in[5];
  const float* fc2_w  = (const float*)d_in[6];
  const float* fc2_b  = (const float*)d_in[7];
  const float* xpw    = (const float*)d_in[8];
  const float* dtw    = (const float*)d_in[9];
  const float* dtb    = (const float*)d_in[10];
  const float* alogs  = (const float*)d_in[11];
  const float* Dsp    = (const float*)d_in[12];
  const float* ln_g   = (const float*)d_in[13];
  const float* ln_b   = (const float*)d_in[14];
  const float* w_out  = (const float*)d_in[15];
  float* out = (float*)d_out;

  char* ws = (char*)d_ws;
  size_t o = 0;
  float* buf_xp = (float*)(ws + o); o += (size_t)MTOT * 256 * 4;   // reused as out_y
  float* buf_z  = (float*)(ws + o); o += (size_t)MTOT * 256 * 4;
  float* xs     = (float*)(ws + o); o += (size_t)MTOT * 256 * 4;   // [B,4,64,L]
  float* xdbl   = (float*)(ws + o); o += (size_t)BATCH * 4 * CDBL * LL * 4; // reused as yln
  float* zz     = (float*)(ws + o); o += 8 * 256 * 4;
  float* fc2o   = (float*)(ws + o); o += 8 * 256 * 4;
  _Float16* x16    = (_Float16*)(ws + o); o += (size_t)MTOT * 256 * 2;
  _Float16* w_in16 = (_Float16*)(ws + o); o += (size_t)512 * 256 * 2;
  _Float16* w_out16= (_Float16*)(ws + o); o += (size_t)256 * 256 * 2;
  _Float16* xpw16  = (_Float16*)(ws + o); o += (size_t)KDIR * CDBL * DGRP * 2;
  float* out_y  = buf_xp;              // xp dead once xs is built
  _Float16* yln = (_Float16*)xdbl;     // x_dbl dead once scan is done

  k_zero<<<8, 256, 0, stream>>>(zz);
  k_cvt<<<(MTOT * 256) / 256, 256, 0, stream>>>(x, x16, MTOT * 256);
  k_cvt<<<(512 * 256) / 256, 256, 0, stream>>>(w_in, w_in16, 512 * 256);
  k_cvt<<<(256 * 256) / 256, 256, 0, stream>>>(w_out, w_out16, 256 * 256);
  k_cvt<<<(KDIR * CDBL * DGRP + 255) / 256, 256, 0, stream>>>(xpw, xpw16, KDIR * CDBL * DGRP);
  k_gemm_inproj<<<8192 / 8, 256, 0, stream>>>(x16, w_in16, buf_xp, buf_z);
  k_elem<<<BATCH * 256, 256, 0, stream>>>(buf_xp, buf_z, conv_w, conv_b, xs, zz);
  k_se<<<BATCH, 256, 0, stream>>>(zz, fc1_w, fc1_b, fc2_w, fc2_b, fc2o);
  k_gemm_xdbl<<<8192 / 8, 256, 0, stream>>>(xs, xpw16, xdbl);
  k_scan<<<128, 256, 0, stream>>>(xs, xdbl, dtw, dtb, alogs, Dsp, out_y);
  k_ln<<<MTOT, 256, 0, stream>>>(out_y, fc2o, buf_z, ln_g, ln_b, yln);
  k_gemm_out<<<4096 / 8, 256, 0, stream>>>(yln, w_out16, out);
}
